// TransformerBlock_88390426952039
// MI455X (gfx1250) — matrix operand order repaired and checked by reading, not compiled, not run on hardware
//
#include <hip/hip_runtime.h>
#include <math.h>

typedef __attribute__((ext_vector_type(16))) _Float16 v16h;
typedef __attribute__((ext_vector_type(8)))  _Float16 v8h;
typedef __attribute__((ext_vector_type(8)))  float    v8f;

#define T_TOK 16384   // B*S = 64*256
#define DMODEL 384

__device__ __forceinline__ v16h mk16(v8h lo, v8h hi) {
  v16h r;
#pragma unroll
  for (int i = 0; i < 8; ++i) { r[i] = lo[i]; r[i + 8] = hi[i]; }
  return r;
}

__device__ __forceinline__ v8f wmma32(v16h a, v16h b, v8f c) {
  return __builtin_amdgcn_wmma_f32_16x16x32_f16(false, a, false, b, (short)0, c,
                                                false, false);
}

// ---- VALU lane permutes (no LDS): v_permlane16_b32 / v_permlanex16_b32 ----
__device__ __forceinline__ float pl16(float v, unsigned s0, unsigned s1) {
  return __uint_as_float(__builtin_amdgcn_permlane16(
      __float_as_uint(v), __float_as_uint(v), s0, s1, false, false));
}
__device__ __forceinline__ float plx16(float v) {  // swap 16-lane halves
  return __uint_as_float(__builtin_amdgcn_permlanex16(
      __float_as_uint(v), __float_as_uint(v), 0x76543210u, 0xFEDCBA98u, false, false));
}
// sel[i] = i ^ m nibble patterns
#define XOR1_LO 0x67452301u
#define XOR1_HI 0xEFCDAB89u
#define XOR2_LO 0x54761032u
#define XOR2_HI 0xDCFE98BAu
#define XOR4_LO 0x32107654u
#define XOR4_HI 0xBA98FEDCu
#define XOR8_LO 0xFEDCBA98u
#define XOR8_HI 0x76543210u

__device__ __forceinline__ float rowmax16(float v) {  // max over 16-lane row
  v = fmaxf(v, pl16(v, XOR1_LO, XOR1_HI));
  v = fmaxf(v, pl16(v, XOR2_LO, XOR2_HI));
  v = fmaxf(v, pl16(v, XOR4_LO, XOR4_HI));
  v = fmaxf(v, pl16(v, XOR8_LO, XOR8_HI));
  return v;
}
__device__ __forceinline__ float sum32(float v) {  // sum over full wave32
  v += pl16(v, XOR1_LO, XOR1_HI);
  v += pl16(v, XOR2_LO, XOR2_HI);
  v += pl16(v, XOR4_LO, XOR4_HI);
  v += pl16(v, XOR8_LO, XOR8_HI);
  v += plx16(v);
  return v;
}

// async copy of 16 bytes global -> LDS (CDNA5, tracked by ASYNCcnt)
__device__ __forceinline__ void async_g2l_b128(unsigned lds_off, const void* gptr) {
  asm volatile("global_load_async_to_lds_b128 %0, %1, off"
               :: "v"(lds_off), "v"((unsigned long long)(size_t)gptr)
               : "memory");
}
__device__ __forceinline__ void wait_async0() {
  asm volatile("s_wait_asynccnt 0x0" ::: "memory");
}

// ---------------- weight conversion ----------------
__global__ void pack_qkv_w(const float* __restrict__ wq, const float* __restrict__ wk,
                           const float* __restrict__ wv, _Float16* __restrict__ out) {
  int i = blockIdx.x * 256 + threadIdx.x;
  if (i >= 384 * 1152) return;
  int r = i / 1152, c = i % 1152;
  float v = (c < 384) ? wq[r * 384 + c]
                      : (c < 768 ? wk[r * 384 + (c - 384)] : wv[r * 384 + (c - 768)]);
  out[i] = (_Float16)v;
}

__global__ void cvt_f16(const float* __restrict__ src, _Float16* __restrict__ dst, int n) {
  int i = blockIdx.x * 256 + threadIdx.x;
  if (i < n) dst[i] = (_Float16)src[i];
}

// ---------------- layernorm: one wave32 per token ----------------
__global__ __launch_bounds__(256) void ln_f16(const float* __restrict__ x,
                                              const float* __restrict__ g,
                                              const float* __restrict__ b,
                                              _Float16* __restrict__ out) {
  int lane = threadIdx.x & 31, wave = threadIdx.x >> 5;
  int token = blockIdx.x * 8 + wave;
  const float* xr = x + (size_t)token * DMODEL;
  float vals[12], s = 0.f, ss = 0.f;
#pragma unroll
  for (int i = 0; i < 12; ++i) {
    float v = xr[lane + 32 * i];
    vals[i] = v; s += v; ss += v * v;
  }
  s = sum32(s);
  ss = sum32(ss);
  float mean = s * (1.f / DMODEL);
  float var = ss * (1.f / DMODEL) - mean * mean;
  float rstd = rsqrtf(var + 1e-5f);
  _Float16* orow = out + (size_t)token * DMODEL;
#pragma unroll
  for (int i = 0; i < 12; ++i) {
    int c = lane + 32 * i;
    orow[c] = (_Float16)((vals[i] - mean) * rstd * g[c] + b[c]);
  }
}

// ---------------- generic f16 WMMA GEMM: C[M,N] = A[M,K]*B[K,N] ----------------
// Block tile 128x128, 8 waves (2 M x 4 N), wave tile 64x32 (4x2 WMMA frags).
// Double-buffered LDS; A tile via global_load_async_to_lds_b128 (ASYNCcnt),
// B tile transposed through registers so both fragment types are contiguous 16B
// LDS loads matching the CDNA5 16-bit A/B VGPR layouts.
template <bool RELU, bool OUTF16>
__global__ __launch_bounds__(256) void gemm_wmma(
    const _Float16* __restrict__ A, const _Float16* __restrict__ B,
    const float* __restrict__ bias, const float* __restrict__ resid,
    void* __restrict__ outv, int M, int N, int K) {
  __shared__ _Float16 As[2][128][32];
  __shared__ _Float16 Bl[2][128][32];
  int tid = threadIdx.x;
  int lane = tid & 31, wave = tid >> 5;
  int waveM = wave >> 2, waveN = wave & 3;
  int m0 = blockIdx.y * 128, n0 = blockIdx.x * 128;
  int halfsel = (lane >= 16) ? 1 : 0;
  int l15 = lane & 15;

  v8f acc[4][2];
#pragma unroll
  for (int i = 0; i < 4; ++i)
#pragma unroll
    for (int j = 0; j < 2; ++j) acc[i][j] = (v8f)0.f;

  int arow = tid >> 1, acol = (tid & 1) * 16;  // A tile: 128 rows x 32 cols
  int bk = tid >> 3, bc = (tid & 7) * 16;      // B tile: 32 rows x 128 cols

  auto load_tile = [&](int kt, int buf) {
    unsigned aoff = (unsigned)(size_t)&As[buf][arow][acol];
    const _Float16* ap = A + (size_t)(m0 + arow) * K + kt + acol;
    async_g2l_b128(aoff, ap);
    async_g2l_b128(aoff + 16u, ap + 8);
    const _Float16* bp = B + (size_t)(kt + bk) * N + n0 + bc;
    v8h b0 = *(const v8h*)bp;
    v8h b1 = *(const v8h*)(bp + 8);
#pragma unroll
    for (int i = 0; i < 8; ++i) {
      Bl[buf][bc + i][bk]     = b0[i];
      Bl[buf][bc + 8 + i][bk] = b1[i];
    }
  };

  load_tile(0, 0);
  wait_async0();
  __syncthreads();

  int nk = K >> 5;
  for (int t = 0; t < nk; ++t) {
    int buf = t & 1;
    if (t + 1 < nk) load_tile((t + 1) << 5, buf ^ 1);

    v16h af[4], bf[2];
#pragma unroll
    for (int mt = 0; mt < 4; ++mt) {
      int r = waveM * 64 + mt * 16 + l15;
      af[mt] = mk16(*(v8h*)&As[buf][r][halfsel * 8],
                    *(v8h*)&As[buf][r][halfsel * 8 + 16]);
    }
#pragma unroll
    for (int nt = 0; nt < 2; ++nt) {
      int c = waveN * 32 + nt * 16 + l15;
      bf[nt] = mk16(*(v8h*)&Bl[buf][c][halfsel * 8],
                    *(v8h*)&Bl[buf][c][halfsel * 8 + 16]);
    }
#pragma unroll
    for (int mt = 0; mt < 4; ++mt)
#pragma unroll
      for (int nt = 0; nt < 2; ++nt) acc[mt][nt] = wmma32(af[mt], bf[nt], acc[mt][nt]);

    wait_async0();
    __syncthreads();
  }

#pragma unroll
  for (int mt = 0; mt < 4; ++mt) {
#pragma unroll
    for (int nt = 0; nt < 2; ++nt) {
      int col = n0 + waveN * 32 + nt * 16 + l15;
      float bv = bias ? bias[col] : 0.f;
#pragma unroll
      for (int j = 0; j < 8; ++j) {
        int row = m0 + waveM * 64 + mt * 16 + j + halfsel * 8;
        float v = acc[mt][nt][j] + bv;
        if (RELU) v = v > 0.f ? v : 0.f;
        if (resid) v += resid[(size_t)row * N + col];
        if constexpr (OUTF16)
          ((_Float16*)outv)[(size_t)row * N + col] = (_Float16)v;
        else
          ((float*)outv)[(size_t)row * N + col] = v;
      }
    }
  }
}

// ---------------- causal flash attention per (head, batch) ----------------
// One 32-key block step; MASKED only for the diagonal block. Row max via
// v_permlane16; row sum via WMMA against an all-ones B fragment.
template <bool MASKED>
__device__ __forceinline__ void attn_block(
    int key0, int r0, int head, const _Float16* base,
    int halfsel, int l15, v16h ones,
    const v16h (*qf)[2], _Float16 (*Vtt)[264], _Float16 (*pl)[32],
    v8f (*acc)[4], float (*mrow)[8], float (*lrow)[8]) {
  // K^T B-fragments straight from global: B(k,n) = K[key0+nt*16+n][kc*32+k]
  v16h kf[2][2];
#pragma unroll
  for (int nt = 0; nt < 2; ++nt) {
    const _Float16* kr = base + (size_t)(key0 + nt * 16 + l15) * 1152 + 384 + head * 64;
#pragma unroll
    for (int kc = 0; kc < 2; ++kc) {
      int k0 = kc * 32 + halfsel * 8;
      kf[nt][kc] = mk16(*(const v8h*)(kr + k0), *(const v8h*)(kr + k0 + 16));
    }
  }
  v8f st[2][2];
#pragma unroll
  for (int mt = 0; mt < 2; ++mt)
#pragma unroll
    for (int nt = 0; nt < 2; ++nt) {
      v8f c = (v8f)0.f;
      c = wmma32(qf[mt][0], kf[nt][0], c);
      c = wmma32(qf[mt][1], kf[nt][1], c);
      st[mt][nt] = c;
    }

  // scale, (diagonal-only) causal mask, running-max update, P = exp(s - m)
  float carr[2][8];
#pragma unroll
  for (int mt = 0; mt < 2; ++mt) {
#pragma unroll
    for (int j = 0; j < 8; ++j) {
      float s0 = st[mt][0][j] * 0.125f;
      float s1 = st[mt][1][j] * 0.125f;
      if constexpr (MASKED) {
        int q = r0 + mt * 16 + j + halfsel * 8;
        if (key0 + l15 > q) s0 = -INFINITY;
        if (key0 + 16 + l15 > q) s1 = -INFINITY;
      }
      float mx = rowmax16(fmaxf(s0, s1));
      float mnew = fmaxf(mrow[mt][j], mx);
      float corr = __expf(mrow[mt][j] - mnew);
      carr[mt][j] = corr;
      mrow[mt][j] = mnew;
      st[mt][0][j] = __expf(s0 - mnew);
      st[mt][1][j] = __expf(s1 - mnew);
    }
  }

  // stage P (C-layout f32 -> f16 A-layout) via wave-private LDS
#pragma unroll
  for (int mt = 0; mt < 2; ++mt)
#pragma unroll
    for (int nt = 0; nt < 2; ++nt)
#pragma unroll
      for (int j = 0; j < 8; ++j)
        pl[mt * 16 + j + halfsel * 8][nt * 16 + l15] = (_Float16)st[mt][nt][j];

  // V B-fragments: B(k,n) = V[key0+k][nt*16+n] -> contiguous reads of Vtt
  v16h vf[4];
#pragma unroll
  for (int nt = 0; nt < 4; ++nt) {
    int d = nt * 16 + l15;
    int s0i = key0 + halfsel * 8;
    vf[nt] = mk16(*(v8h*)&Vtt[d][s0i], *(v8h*)&Vtt[d][s0i + 16]);
  }
#pragma unroll
  for (int mt = 0; mt < 2; ++mt) {
    v16h pf = mk16(*(v8h*)&pl[mt * 16 + l15][halfsel * 8],
                   *(v8h*)&pl[mt * 16 + l15][halfsel * 8 + 16]);
    // row sums of P via WMMA against ones: lsum[j] = sum_k P[row][k]
    v8f lsum = wmma32(pf, ones, (v8f)0.f);
#pragma unroll
    for (int j = 0; j < 8; ++j)
      lrow[mt][j] = lrow[mt][j] * carr[mt][j] + lsum[j];
#pragma unroll
    for (int nt = 0; nt < 4; ++nt)
#pragma unroll
      for (int j = 0; j < 8; ++j) acc[mt][nt][j] *= carr[mt][j];
#pragma unroll
    for (int nt = 0; nt < 4; ++nt) acc[mt][nt] = wmma32(pf, vf[nt], acc[mt][nt]);
  }
}

// qkv: [B*S, 1152] f16 rows = [q(384)|k(384)|v(384)], per-head 64 cols.
// 8 waves; wave w owns query rows w*32..w*32+31.
__global__ __launch_bounds__(256) void attn_wmma(const _Float16* __restrict__ qkv,
                                                 _Float16* __restrict__ ctx) {
  __shared__ _Float16 Vtt[64][264];   // V transposed [dim][seq] (+pad)
  __shared__ _Float16 Pl[8][32][32];  // per-wave P re-layout staging

  int head = blockIdx.x, batch = blockIdx.y;
  int tid = threadIdx.x, lane = tid & 31, wave = tid >> 5;
  int halfsel = (lane >= 16) ? 1 : 0;
  int l15 = lane & 15;
  const _Float16* base = qkv + (size_t)batch * 256 * 1152;

  {  // stage V transposed: thread t handles seq position t
    int s = tid;
    const _Float16* vr = base + (size_t)s * 1152 + 768 + head * 64;
#pragma unroll
    for (int d0 = 0; d0 < 64; d0 += 8) {
      v8h vv = *(const v8h*)(vr + d0);
#pragma unroll
      for (int i = 0; i < 8; ++i) Vtt[d0 + i][s] = vv[i];
    }
  }
  __syncthreads();

  int r0 = wave * 32;
  v16h qf[2][2];
#pragma unroll
  for (int mt = 0; mt < 2; ++mt) {
    const _Float16* qr = base + (size_t)(r0 + mt * 16 + l15) * 1152 + head * 64;
#pragma unroll
    for (int kc = 0; kc < 2; ++kc) {
      int k0 = kc * 32 + halfsel * 8;
      qf[mt][kc] = mk16(*(const v8h*)(qr + k0), *(const v8h*)(qr + k0 + 16));
    }
  }

  v16h ones;
#pragma unroll
  for (int i = 0; i < 16; ++i) ones[i] = (_Float16)1.f;

  v8f acc[2][4];
#pragma unroll
  for (int i = 0; i < 2; ++i)
#pragma unroll
    for (int j = 0; j < 4; ++j) acc[i][j] = (v8f)0.f;
  float mrow[2][8], lrow[2][8];
#pragma unroll
  for (int i = 0; i < 2; ++i)
#pragma unroll
    for (int j = 0; j < 8; ++j) { mrow[i][j] = -INFINITY; lrow[i][j] = 0.f; }

  int nfull = r0 >> 5;  // key blocks strictly below the diagonal: no masking
  for (int kb = 0; kb < nfull; ++kb)
    attn_block<false>(kb * 32, r0, head, base, halfsel, l15, ones, qf, Vtt,
                      Pl[wave], acc, mrow, lrow);
  attn_block<true>(nfull * 32, r0, head, base, halfsel, l15, ones, qf, Vtt,
                   Pl[wave], acc, mrow, lrow);

  // normalize and write ctx (heads concatenated)
#pragma unroll
  for (int mt = 0; mt < 2; ++mt)
#pragma unroll
    for (int nt = 0; nt < 4; ++nt)
#pragma unroll
      for (int j = 0; j < 8; ++j) {
        int q = r0 + mt * 16 + j + halfsel * 8;
        float v = acc[mt][nt][j] / lrow[mt][j];
        ctx[(size_t)(batch * 256 + q) * 384 + head * 64 + nt * 16 + l15] = (_Float16)v;
      }
}

// ---------------- host ----------------
extern "C" void kernel_launch(void* const* d_in, const int* in_sizes, int n_in,
                              void* d_out, int out_size, void* d_ws, size_t ws_size,
                              hipStream_t stream) {
  (void)in_sizes; (void)n_in; (void)out_size; (void)ws_size;
  const float* x  = (const float*)d_in[0];
  const float* wq = (const float*)d_in[1];
  const float* wk = (const float*)d_in[2];
  const float* wv = (const float*)d_in[3];
  const float* wo = (const float*)d_in[4];
  const float* bo = (const float*)d_in[5];
  const float* w1 = (const float*)d_in[6];
  const float* b1 = (const float*)d_in[7];
  const float* w2 = (const float*)d_in[8];
  const float* b2 = (const float*)d_in[9];
  const float* g1  = (const float*)d_in[10];
  const float* be1 = (const float*)d_in[11];
  const float* g2  = (const float*)d_in[12];
  const float* be2 = (const float*)d_in[13];
  float* out = (float*)d_out;

  char* ws = (char*)d_ws;
  size_t off = 0;
  auto alloc = [&](size_t bytes) -> void* {
    void* p = ws + off;
    off = (off + bytes + 255) & ~(size_t)255;
    return p;
  };
  _Float16* wqkv16 = (_Float16*)alloc((size_t)384 * 1152 * 2);
  _Float16* wo16   = (_Float16*)alloc((size_t)384 * 384 * 2);
  _Float16* w116   = (_Float16*)alloc((size_t)384 * 1536 * 2);
  _Float16* w216   = (_Float16*)alloc((size_t)1536 * 384 * 2);
  _Float16* h16    = (_Float16*)alloc((size_t)T_TOK * 384 * 2);
  _Float16* qkv16  = (_Float16*)alloc((size_t)T_TOK * 1152 * 2);
  _Float16* ctx16  = (_Float16*)alloc((size_t)T_TOK * 384 * 2);
  float*    x1     = (float*)alloc((size_t)T_TOK * 384 * 4);
  _Float16* h216   = (_Float16*)alloc((size_t)T_TOK * 384 * 2);
  _Float16* ff116  = (_Float16*)alloc((size_t)T_TOK * 1536 * 2);

  pack_qkv_w<<<(384 * 1152 + 255) / 256, 256, 0, stream>>>(wq, wk, wv, wqkv16);
  cvt_f16<<<(384 * 384 + 255) / 256, 256, 0, stream>>>(wo, wo16, 384 * 384);
  cvt_f16<<<(384 * 1536 + 255) / 256, 256, 0, stream>>>(w1, w116, 384 * 1536);
  cvt_f16<<<(1536 * 384 + 255) / 256, 256, 0, stream>>>(w2, w216, 1536 * 384);

  ln_f16<<<T_TOK / 8, 256, 0, stream>>>(x, g1, be1, h16);
  gemm_wmma<false, true><<<dim3(9, 128), 256, 0, stream>>>(
      h16, wqkv16, nullptr, nullptr, qkv16, T_TOK, 1152, 384);
  attn_wmma<<<dim3(6, 64), 256, 0, stream>>>(qkv16, ctx16);
  gemm_wmma<false, false><<<dim3(3, 128), 256, 0, stream>>>(
      ctx16, wo16, bo, x, x1, T_TOK, 384, 384);
  ln_f16<<<T_TOK / 8, 256, 0, stream>>>(x1, g2, be2, h216);
  gemm_wmma<true, true><<<dim3(12, 128), 256, 0, stream>>>(
      h216, w116, b1, nullptr, ff116, T_TOK, 1536, 384);
  gemm_wmma<false, false><<<dim3(3, 128), 256, 0, stream>>>(
      ff116, w216, b2, x1, out, T_TOK, 384, 1536);
}
